// VectorQuantizer_69217692943004
// MI455X (gfx1250) — compile-verified, matrix operand order checked by reading, not run on hardware
//
#include <hip/hip_runtime.h>
#include <hip/hip_bf16.h>
#include <cstdint>
#include <cstddef>

typedef __attribute__((ext_vector_type(16))) __bf16 v16bf;
typedef __attribute__((ext_vector_type(8)))  float  v8f;

#define DIM     512
#define KCB     8192
#define NROWS   16384
#define NTILES  (KCB / 16)    // 512 column tiles
#define CHUNKS  (DIM / 32)    // 16 k-chunks per WMMA sweep

union V16 {
    v16bf    bf;
    uint32_t u[8];
    uint4    q[2];
};

__device__ __forceinline__ uint16_t f2bf(float f) {
    uint32_t u = __float_as_uint(f);
    u += 0x7FFFu + ((u >> 16) & 1u);     // round to nearest even
    return (uint16_t)(u >> 16);
}
__device__ __forceinline__ float bf2f(uint16_t h) {
    return __uint_as_float(((uint32_t)h) << 16);
}

// CDNA5 async global->LDS copy (zero VGPR staging, tracked by ASYNCcnt).
// GV mode: 64-bit global address in VGPR pair, per-lane LDS byte address in
// a VGPR (generic LDS pointers carry the LDS offset in their low 32 bits).
__device__ __forceinline__ void async_ld_b128(void* ldsPtr, const void* gPtr) {
    uint32_t l = (uint32_t)(uintptr_t)ldsPtr;
    asm volatile("global_load_async_to_lds_b128 %0, %1, off"
                 :: "v"(l), "v"(gPtr)
                 : "memory");
}
__device__ __forceinline__ void wait_async0() {
    asm volatile("s_wait_asynccnt 0" ::: "memory");
}

// ---------------------------------------------------------------------------
// Kernel 1: codebook -> bf16 hi/lo planes + e2[k] = ||e_k||^2
// One wave per codebook row.
// ---------------------------------------------------------------------------
__global__ __launch_bounds__(256) void vq_prep_codebook(
    const float* __restrict__ cb,
    uint16_t* __restrict__ ehi, uint16_t* __restrict__ elo,
    float* __restrict__ e2)
{
    const int wave = blockIdx.x * (blockDim.x >> 5) + (threadIdx.x >> 5);
    const int lane = threadIdx.x & 31;
    if (wave >= KCB) return;

    const float* src = cb + (size_t)wave * DIM;
    uint16_t* dh = ehi + (size_t)wave * DIM;
    uint16_t* dl = elo + (size_t)wave * DIM;

    float ss = 0.f;
#pragma unroll
    for (int i = 0; i < DIM / 32; ++i) {
        const int d = i * 32 + lane;
        const float f = src[d];
        const uint16_t h = f2bf(f);
        const uint16_t l = f2bf(f - bf2f(h));
        dh[d] = h;
        dl[d] = l;
        ss += f * f;
    }
#pragma unroll
    for (int m = 16; m >= 1; m >>= 1) ss += __shfl_xor(ss, m, 32);
    if (lane == 0) e2[wave] = ss;
}

// ---------------------------------------------------------------------------
// Kernel 2: main VQ GEMM + argmin.
// Block = 128 threads (4 waves). Each wave owns 16 rows of x, held in
// registers in native WMMA A-layout (hi/lo bf16). Codebook streamed in
// 16-entry tiles through double-buffered LDS, refilled with CDNA5
// global_load_async_to_lds_b128 (no VGPR staging -> no spills).
// ---------------------------------------------------------------------------
__global__ __launch_bounds__(128, 1) void vq_main(
    const float* __restrict__ x, const float* __restrict__ cb,
    const uint16_t* __restrict__ ehi, const uint16_t* __restrict__ elo,
    const float* __restrict__ e2,
    float* __restrict__ qOut, float* __restrict__ idxOut,
    float* __restrict__ minOut)
{
    __shared__ __align__(16) uint16_t ldsHi[2][16 * DIM];  // 2 x 16 KB
    __shared__ __align__(16) uint16_t ldsLo[2][16 * DIM];  // 2 x 16 KB

    const int tid     = threadIdx.x;
    const int lane    = tid & 31;
    const int waveId  = tid >> 5;
    const int laneHi  = lane >> 4;   // 0: K 0..7 / 16..23 ; 1: K 8..15 / 24..31
    const int laneLo  = lane & 15;
    const int rowBase = (blockIdx.x * 4 + waveId) * 16;

    // ---- Load this wave's 16x512 x-tile into A-layout registers ----------
    V16 ahi[CHUNKS], alo[CHUNKS];
    float x2part = 0.f;
    const float* xrow = x + (size_t)(rowBase + laneLo) * DIM + laneHi * 8;
#pragma unroll
    for (int c = 0; c < CHUNKS; ++c) {
        const float* p = xrow + c * 32;
        float f[16];
#pragma unroll
        for (int i = 0; i < 8; ++i) f[i] = p[i];
#pragma unroll
        for (int i = 0; i < 8; ++i) f[8 + i] = p[16 + i];
#pragma unroll
        for (int i = 0; i < 16; ++i) x2part += f[i] * f[i];
#pragma unroll
        for (int i = 0; i < 8; ++i) {
            const uint16_t h0 = f2bf(f[2 * i]);
            const uint16_t h1 = f2bf(f[2 * i + 1]);
            const uint16_t l0 = f2bf(f[2 * i]     - bf2f(h0));
            const uint16_t l1 = f2bf(f[2 * i + 1] - bf2f(h1));
            ahi[c].u[i] = (uint32_t)h0 | ((uint32_t)h1 << 16);
            alo[c].u[i] = (uint32_t)l0 | ((uint32_t)l1 << 16);
        }
    }
    // Row r's elements are covered exactly by lanes r and r+16.
    const float x2row = x2part + __shfl_xor(x2part, 16, 32);

    // ---- Stage codebook tile 0 into LDS buffer 0 (async) -----------------
    // 16 KB per plane; 128 threads x 8 x b128 per plane.
    {
        const char* sh = (const char*)ehi + tid * 16;
        const char* sl = (const char*)elo + tid * 16;
        char* dh = (char*)&ldsHi[0][0] + tid * 16;
        char* dl = (char*)&ldsLo[0][0] + tid * 16;
#pragma unroll
        for (int k = 0; k < 8; ++k) {
            async_ld_b128(dh + k * 2048, sh + k * 2048);
            async_ld_b128(dl + k * 2048, sl + k * 2048);
        }
    }
    wait_async0();
    __syncthreads();

    float minv[8];
    int   mini[8];
#pragma unroll
    for (int j = 0; j < 8; ++j) { minv[j] = 3.0e38f; mini[j] = 0; }

#pragma unroll 1
    for (int t = 0; t < NTILES; ++t) {
        const int buf = t & 1;
        const bool more = (t + 1) < NTILES;

        // Kick off async refill of the other buffer (overlaps the WMMAs).
        if (more) {
            const char* sh = (const char*)ehi + (size_t)(t + 1) * 16384 + tid * 16;
            const char* sl = (const char*)elo + (size_t)(t + 1) * 16384 + tid * 16;
            char* dh = (char*)&ldsHi[buf ^ 1][0] + tid * 16;
            char* dl = (char*)&ldsLo[buf ^ 1][0] + tid * 16;
#pragma unroll
            for (int k = 0; k < 8; ++k) {
                async_ld_b128(dh + k * 2048, sh + k * 2048);
                async_ld_b128(dl + k * 2048, sl + k * 2048);
            }
        }
        const float e2v = e2[t * 16 + laneLo];

        // ---- 16x16 dot-product tile: 3 bf16 WMMAs per K=32 chunk ---------
        v8f acc = {};
        const uint16_t* bh = &ldsHi[buf][laneLo * DIM + laneHi * 8];
        const uint16_t* bl = &ldsLo[buf][laneLo * DIM + laneHi * 8];
#pragma unroll
        for (int c = 0; c < CHUNKS; ++c) {
            V16 Bh, Bl;
            Bh.q[0] = *(const uint4*)(bh + c * 32);
            Bh.q[1] = *(const uint4*)(bh + c * 32 + 16);
            Bl.q[0] = *(const uint4*)(bl + c * 32);
            Bl.q[1] = *(const uint4*)(bl + c * 32 + 16);
            acc = __builtin_amdgcn_wmma_f32_16x16x32_bf16(
                false, ahi[c].bf, false, Bh.bf, (short)0, acc, false, false);
            acc = __builtin_amdgcn_wmma_f32_16x16x32_bf16(
                false, ahi[c].bf, false, Bl.bf, (short)0, acc, false, false);
            acc = __builtin_amdgcn_wmma_f32_16x16x32_bf16(
                false, alo[c].bf, false, Bh.bf, (short)0, acc, false, false);
        }

        // Partial distance p = ||e||^2 - 2 x.e  (x^2 folded in at the end).
        const int col = t * 16 + laneLo;
#pragma unroll
        for (int j = 0; j < 8; ++j) {
            const float p = e2v - 2.0f * acc[j];
            if (p < minv[j]) { minv[j] = p; mini[j] = col; }
        }

        if (more) {
            wait_async0();       // this wave's async writes have landed in LDS
            __syncthreads();     // all waves' writes visible, all readers done
        }
    }

    // ---- Per-row argmin reduce + outputs ---------------------------------
    // Acc slot j: lanes 0..15 hold row rowBase+j, lanes 16..31 row rowBase+j+8;
    // columns striped over the 16 lanes of each half.
#pragma unroll
    for (int j = 0; j < 8; ++j) {
        float v = minv[j];
        int   ix = mini[j];
#pragma unroll
        for (int m = 1; m < 16; m <<= 1) {
            const float ov = __shfl_xor(v, m, 32);
            const int   oi = __shfl_xor(ix, m, 32);
            if (ov < v || (ov == v && oi < ix)) { v = ov; ix = oi; }
        }
        const float x2a = __shfl(x2row, j, 32);       // row rowBase+j
        const float x2b = __shfl(x2row, j + 8, 32);   // row rowBase+j+8
        const int b0  = __shfl(ix, 0, 32);
        const int b16 = __shfl(ix, 16, 32);

        if (lane == 0) {
            const int r = rowBase + j;
            idxOut[r] = (float)ix;
            minOut[r] = v + x2a;
        }
        if (lane == 16) {
            const int r = rowBase + j + 8;
            idxOut[r] = (float)ix;
            minOut[r] = v + x2b;
        }

        // quantized_st row = codebook[argmin] (fp32 gather, whole wave copies)
        const float4* s0 = (const float4*)(cb + (size_t)b0 * DIM);
        float4*       d0 = (float4*)(qOut + (size_t)(rowBase + j) * DIM);
        const float4* s1 = (const float4*)(cb + (size_t)b16 * DIM);
        float4*       d1 = (float4*)(qOut + (size_t)(rowBase + j + 8) * DIM);
#pragma unroll
        for (int tt = 0; tt < 4; ++tt) {
            d0[lane + tt * 32] = s0[lane + tt * 32];
            d1[lane + tt * 32] = s1[lane + tt * 32];
        }
    }
}

// ---------------------------------------------------------------------------
// Kernel 3: deterministic loss reduction.
// loss_vq == loss_commit == sum of min distances S; loss = 0.25*S + S.
// ---------------------------------------------------------------------------
__global__ __launch_bounds__(256) void vq_finalize(
    const float* __restrict__ minOut,
    float* __restrict__ lossOut, float* __restrict__ commitOut)
{
    __shared__ float red[256];
    float s = 0.f;
    for (int i = threadIdx.x; i < NROWS; i += 256) s += minOut[i];
    red[threadIdx.x] = s;
    __syncthreads();
    for (int off = 128; off > 0; off >>= 1) {
        if (threadIdx.x < off) red[threadIdx.x] += red[threadIdx.x + off];
        __syncthreads();
    }
    if (threadIdx.x == 0) {
        const float S = red[0];
        *lossOut   = 1.25f * S;
        *commitOut = S;
    }
}

// ---------------------------------------------------------------------------
extern "C" void kernel_launch(void* const* d_in, const int* in_sizes, int n_in,
                              void* d_out, int out_size, void* d_ws, size_t ws_size,
                              hipStream_t stream)
{
    (void)in_sizes; (void)n_in; (void)out_size; (void)ws_size;

    const float* x  = (const float*)d_in[0];   // [32,512,512]
    const float* cb = (const float*)d_in[1];   // [8192,512]
    float* out = (float*)d_out;

    // Workspace: bf16 hi plane (8 MB) | bf16 lo plane (8 MB) | e2 (32 KB)
    char* ws = (char*)d_ws;
    uint16_t* ehi = (uint16_t*)ws;
    uint16_t* elo = (uint16_t*)(ws + (size_t)KCB * DIM * 2);
    float*    e2  = (float*)(ws + (size_t)KCB * DIM * 4);

    // Output layout (concatenated, reference return order):
    float* qOut   = out;                          // 32*512*512
    float* lossO  = out + (size_t)NROWS * DIM;    // 1
    float* idxO   = lossO + 1;                    // 16384
    float* minO   = idxO + NROWS;                 // 16384
    float* commit = minO + NROWS;                 // 1

    vq_prep_codebook<<<KCB / 8, 256, 0, stream>>>(cb, ehi, elo, e2);
    vq_main<<<NROWS / 64, 128, 0, stream>>>(x, cb, ehi, elo, e2, qOut, idxO, minO);
    vq_finalize<<<1, 256, 0, stream>>>(minO, lossO, commit);
}
